// RGCN_61057255079944
// MI455X (gfx1250) — matrix operand order repaired and checked by reading, not compiled, not run on hardware
//
#include <hip/hip_runtime.h>

typedef __attribute__((ext_vector_type(16))) _Float16 v16h;
typedef __attribute__((ext_vector_type(8)))  _Float16 v8h;
typedef __attribute__((ext_vector_type(8)))  float    v8f;

#define NUM_RELS    65
#define NUM_GRAPHS  1500
#define NUM_MOTIFS  6001
#define HDIM        128
#define FFN         512
#define MAX_TILES   5760
#define BN_EPS      1e-5f

// ---------------- WMMA helpers (CDNA5 wave32, 16x16x32 f16 -> f32) ----------
__device__ __forceinline__ v16h load_afrag(const _Float16* p) {
  // A fragment: this lane's row, 16 contiguous halves at p
  v8h lo = *(const v8h*)(p);
  v8h hi = *(const v8h*)(p + 8);
  v16h r;
#pragma unroll
  for (int i = 0; i < 8; ++i) { r[i] = lo[i]; r[i + 8] = hi[i]; }
  return r;
}

__device__ __forceinline__ v8f wmma16(v16h a, v16h b, v8f c) {
  return __builtin_amdgcn_wmma_f32_16x16x32_f16(false, a, false, b, (short)0, c,
                                                false, false);
}

// ---------------- utility kernels ------------------------------------------
__global__ void k_zero(float* p, int n) {
  int i = blockIdx.x * blockDim.x + threadIdx.x;
  if (i < n) p[i] = 0.0f;
}

__global__ void k_cvt(const float* s, _Float16* d, int n) {
  int i = blockIdx.x * blockDim.x + threadIdx.x;
  if (i < n) d[i] = (_Float16)s[i];
}

// d[r][o][i] = s[r][i][o]  (store weights as Bt[N][K] for contiguous B frags)
__global__ void k_tcvt(const float* s, _Float16* d, int R, int rows, int cols) {
  int i = blockIdx.x * blockDim.x + threadIdx.x;
  int per = rows * cols;
  if (i >= R * per) return;
  int r = i / per, rem = i - r * per;
  int o = rem / rows, ii = rem - o * rows;
  d[i] = (_Float16)s[r * per + ii * cols + o];
}

// ---------------- edge grouping by relation --------------------------------
__global__ void k_hist(const int* etype, int* cnt, int E) {
  int e = blockIdx.x * blockDim.x + threadIdx.x;
  if (e < E) atomicAdd(&cnt[etype[e]], 1);
}

__global__ void k_scan_tiles(const int* cnt, int* off, int* tileR, int* tileO,
                             int* ntile) {
  if (blockIdx.x != 0 || threadIdx.x != 0) return;
  int acc = 0, idx = 0;
  for (int r = 0; r < NUM_RELS; ++r) {
    off[r] = acc;
    int c = cnt[r];
    int t = (c + 15) >> 4;
    for (int k = 0; k < t; ++k) {
      tileR[idx] = r;
      tileO[idx] = acc + k * 16;
      ++idx;
    }
    acc += c;
  }
  off[NUM_RELS] = acc;
  ntile[0] = idx;
}

__global__ void k_scatter_edges(const int* src, const int* dst, const int* etype,
                                const int* off, int* cursor, int* esrc, int* edst,
                                int E) {
  int e = blockIdx.x * blockDim.x + threadIdx.x;
  if (e >= E) return;
  int r = etype[e];
  int pos = off[r] + atomicAdd(&cursor[r], 1);
  esrc[pos] = src[e];
  edst[pos] = dst[e];
}

// ---------------- message kernel: 16 edges x 128 outputs per block ---------
__global__ void __launch_bounds__(32)
k_msg(const _Float16* __restrict__ xh, const _Float16* __restrict__ Wt,
      const int* __restrict__ esrc, const int* __restrict__ edst,
      const int* __restrict__ off, const int* __restrict__ tileR,
      const int* __restrict__ tileO, const int* __restrict__ ntile,
      float* __restrict__ agg) {
  int tid = blockIdx.x;
  if (tid >= ntile[0]) return;
  int r = tileR[tid];
  int base = tileO[tid];
  int segEnd = off[r + 1];

  int lane = threadIdx.x;
  int mrow = lane & 15;
  int kb = (lane >> 4) * 16;
  int e = base + mrow;
  bool rowValid = (e < segEnd);
  int sn = rowValid ? esrc[e] : 0;

  v16h a[4];
#pragma unroll
  for (int t = 0; t < 4; ++t) {
    if (rowValid) {
      a[t] = load_afrag(xh + (size_t)sn * HDIM + t * 32 + kb);
    } else {
#pragma unroll
      for (int i = 0; i < 16; ++i) a[t][i] = (_Float16)0.0f;
    }
  }

  int col = lane & 15;
  int kg = (lane >> 4) * 16;
  int mb = (lane >> 4) * 8;

  // Hoisted destination rows + validity for this lane's 8 output rows:
  // invariant across the 8 n-tiles (saves 56 global loads / block and
  // shortens the predicated atomic regions).
  float* dadr[8];
  bool dval[8];
#pragma unroll
  for (int j = 0; j < 8; ++j) {
    int ee = base + mb + j;
    dval[j] = (ee < segEnd);
    int dn = dval[j] ? edst[ee] : 0;
    dadr[j] = agg + (size_t)dn * HDIM + col;
  }

  const _Float16* W = Wt + (size_t)r * HDIM * HDIM;
#pragma unroll
  for (int nt = 0; nt < 8; ++nt) {
    v8f c = {};
#pragma unroll
    for (int t = 0; t < 4; ++t) {
      v16h b = *(const v16h*)(W + (size_t)(nt * 16 + col) * HDIM + t * 32 + kg);
      c = wmma16(a[t], b, c);
    }
#pragma unroll
    for (int j = 0; j < 8; ++j) {
      if (dval[j]) atomicAdd(dadr[j] + nt * 16, c[j]);
    }
  }
}

// --------- node transform: h = relu(agg+b+x@loopW) + relu(x@resW^T + resb) --
__global__ void __launch_bounds__(32)
k_node(const _Float16* __restrict__ xh, const _Float16* __restrict__ loopT,
       const _Float16* __restrict__ resT, const float* __restrict__ bias,
       const float* __restrict__ resb, float* __restrict__ buf,
       float* __restrict__ bnsum) {
  int node0 = blockIdx.x * 16;
  int lane = threadIdx.x;
  int mrow = lane & 15;
  int kb = (lane >> 4) * 16;

  v16h a[4];
#pragma unroll
  for (int t = 0; t < 4; ++t)
    a[t] = load_afrag(xh + (size_t)(node0 + mrow) * HDIM + t * 32 + kb);

  int col = lane & 15;
  int kg = (lane >> 4) * 16;
  int mb = (lane >> 4) * 8;

#pragma unroll
  for (int nt = 0; nt < 8; ++nt) {
    v8f cl = {}, cr = {};
#pragma unroll
    for (int t = 0; t < 4; ++t) {
      v16h bl = *(const v16h*)(loopT + (size_t)(nt * 16 + col) * HDIM + t * 32 + kg);
      cl = wmma16(a[t], bl, cl);
      v16h br = *(const v16h*)(resT + (size_t)(nt * 16 + col) * HDIM + t * 32 + kg);
      cr = wmma16(a[t], br, cr);
    }
    int n = nt * 16 + col;
    float bb = bias[n], rb = resb[n];
    float ps = 0.0f, pq = 0.0f;
#pragma unroll
    for (int j = 0; j < 8; ++j) {
      size_t idx = (size_t)(node0 + mb + j) * HDIM + n;
      float h = fmaxf(buf[idx] + bb + cl[j], 0.0f);
      h += fmaxf(cr[j] + rb, 0.0f);
      buf[idx] = h;
      ps += h;
      pq += h * h;
    }
    atomicAdd(&bnsum[n], ps);
    atomicAdd(&bnsum[HDIM + n], pq);
  }
}

__global__ void k_bnfin(const float* sums, const float* g, const float* be,
                        float* sc, int n) {
  int c = threadIdx.x;
  if (c >= HDIM) return;
  float inv = 1.0f / (float)n;
  float mu = sums[c] * inv;
  float var = sums[HDIM + c] * inv - mu * mu;
  float s = g[c] * rsqrtf(var + BN_EPS);
  sc[c] = s;
  sc[HDIM + c] = be[c] - mu * s;
}

__global__ void k_bnapply(float* buf, const float* sc, _Float16* fb, int n) {
  int i = blockIdx.x * blockDim.x + threadIdx.x;
  if (i >= n) return;
  int c = i & (HDIM - 1);
  float v = buf[i] * sc[c] + sc[HDIM + c];
  buf[i] = v;
  fb[i] = (_Float16)v;
}

// ---------------- segment means --------------------------------------------
__global__ void k_segscat(const float* __restrict__ h, const int* __restrict__ gid,
                          const int* __restrict__ mid, float* __restrict__ gsum,
                          float* __restrict__ gcnt, float* __restrict__ msum,
                          float* __restrict__ mcnt, int N) {
  int idx = blockIdx.x * blockDim.x + threadIdx.x;
  if (idx >= N * 32) return;
  int n = idx >> 5;
  int q = (idx & 31) * 4;
  int g = gid[n], m = mid[n];
#pragma unroll
  for (int k = 0; k < 4; ++k) {
    float v = h[(size_t)n * HDIM + q + k];
    atomicAdd(&gsum[(size_t)g * HDIM + q + k], v);
    atomicAdd(&msum[(size_t)m * HDIM + q + k], v);
  }
  if (q == 0) {
    atomicAdd(&gcnt[g], 1.0f);
    atomicAdd(&mcnt[m], 1.0f);
  }
}

__global__ void k_segdiv(float* s, const float* cnt, int rows, int drop,
                         int writeF32, _Float16* outH) {
  int i = blockIdx.x * blockDim.x + threadIdx.x;
  if (i >= rows * HDIM) return;
  int r = i >> 7, c = i & (HDIM - 1);
  int sr = r + drop;
  float v = s[(size_t)sr * HDIM + c] / fmaxf(cnt[sr], 1.0f);
  if (writeF32) s[(size_t)sr * HDIM + c] = v;
  outH[i] = (_Float16)v;
}

// ---------------- generic WMMA GEMM: out = act(A[M,K] @ Bt[N,K]^T + bias) ---
__global__ void __launch_bounds__(32)
k_gemm(const _Float16* __restrict__ A, const _Float16* __restrict__ Bt,
       const float* __restrict__ bias, float* __restrict__ outF,
       _Float16* __restrict__ outH, int M, int K, int N, int relu) {
  int mt = blockIdx.x, nt = blockIdx.y;
  int lane = threadIdx.x;
  int mrow = lane & 15;
  int kb = (lane >> 4) * 16;
  int row = mt * 16 + mrow;
  if (row >= M) row = M - 1;  // clamp for tail tiles (stores are guarded)
  int col = lane & 15;
  int kg = (lane >> 4) * 16;
  int mb = (lane >> 4) * 8;

  v8f c = {};
  for (int kk = 0; kk < K; kk += 32) {
    v16h a = load_afrag(A + (size_t)row * K + kk + kb);
    v16h b = *(const v16h*)(Bt + (size_t)(nt * 16 + col) * K + kk + kg);
    c = wmma16(a, b, c);
  }
  int n = nt * 16 + col;
  float bb = bias ? bias[n] : 0.0f;
#pragma unroll
  for (int j = 0; j < 8; ++j) {
    int r = mt * 16 + mb + j;
    if (r < M) {
      float v = c[j] + bb;
      if (relu) v = fmaxf(v, 0.0f);
      if (outF) outF[(size_t)r * N + n] = v;
      if (outH) outH[(size_t)r * N + n] = (_Float16)v;
    }
  }
}

// ---------------- host launcher --------------------------------------------
static inline void launch_zero(float* p, size_t n, hipStream_t s) {
  k_zero<<<(unsigned)((n + 255) / 256), 256, 0, s>>>(p, (int)n);
}

extern "C" void kernel_launch(void* const* d_in, const int* in_sizes, int n_in,
                              void* d_out, int out_size, void* d_ws, size_t ws_size,
                              hipStream_t stream) {
  const float* node_feats = (const float*)d_in[0];
  const int* src  = (const int*)d_in[1];
  const int* dst  = (const int*)d_in[2];
  const int* ety  = (const int*)d_in[3];
  const int* gid  = (const int*)d_in[4];
  const int* mid  = (const int*)d_in[5];
  const float* W1 = (const float*)d_in[6];
  const float* loopW1 = (const float*)d_in[7];
  const float* b1 = (const float*)d_in[8];
  const float* resW1 = (const float*)d_in[9];
  const float* resb1 = (const float*)d_in[10];
  const float* g1 = (const float*)d_in[11];
  const float* be1 = (const float*)d_in[12];
  const float* W2 = (const float*)d_in[13];
  const float* loopW2 = (const float*)d_in[14];
  const float* b2 = (const float*)d_in[15];
  const float* resW2 = (const float*)d_in[16];
  const float* resb2 = (const float*)d_in[17];
  const float* g2 = (const float*)d_in[18];
  const float* be2 = (const float*)d_in[19];
  const float* feat_W = (const float*)d_in[20];
  const float* feat_b = (const float*)d_in[21];
  const float* mlp_W1 = (const float*)d_in[22];
  const float* mlp_b1 = (const float*)d_in[23];
  const float* mlp_W2 = (const float*)d_in[24];
  const float* mlp_b2 = (const float*)d_in[25];

  const int N = in_sizes[0] / HDIM;   // 30000
  const int E = in_sizes[1];          // 90000

  // ---- workspace layout ----
  char* w = (char*)d_ws;
  size_t off = 0;
  auto alloc = [&](size_t bytes) -> void* {
    void* p = w + off;
    off = (off + bytes + 255) & ~(size_t)255;
    return p;
  };
  int* ecnt   = (int*)alloc(130 * 4);              // [0,65): counts, [65,130): cursors
  int* eoff   = (int*)alloc(66 * 4);
  int* tileR  = (int*)alloc(MAX_TILES * 4);
  int* tileO  = (int*)alloc(MAX_TILES * 4);
  int* ntile  = (int*)alloc(4);
  int* esrc   = (int*)alloc((size_t)E * 4);
  int* edst   = (int*)alloc((size_t)E * 4);
  _Float16* Wt1    = (_Float16*)alloc((size_t)NUM_RELS * HDIM * HDIM * 2);
  _Float16* Wt2    = (_Float16*)alloc((size_t)NUM_RELS * HDIM * HDIM * 2);
  _Float16* loopT1 = (_Float16*)alloc(HDIM * HDIM * 2);
  _Float16* loopT2 = (_Float16*)alloc(HDIM * HDIM * 2);
  _Float16* resH1  = (_Float16*)alloc(HDIM * HDIM * 2);
  _Float16* resH2  = (_Float16*)alloc(HDIM * HDIM * 2);
  _Float16* featWh = (_Float16*)alloc((size_t)FFN * HDIM * 2);
  _Float16* mlpW1h = (_Float16*)alloc((size_t)FFN * FFN * 2);
  _Float16* mlpW2h = (_Float16*)alloc((size_t)(FFN / 2) * FFN * 2);
  float* bnsum = (float*)alloc(256 * 4);
  float* bnsc  = (float*)alloc(256 * 4);
  float* buf   = (float*)alloc((size_t)N * HDIM * 4);
  _Float16* fbuf = (_Float16*)alloc((size_t)N * HDIM * 2);
  // msum | mcnt | gcnt contiguous so one zero pass covers all
  size_t msum_elts = (size_t)NUM_MOTIFS * HDIM + NUM_MOTIFS + NUM_GRAPHS;
  float* msum = (float*)alloc(msum_elts * 4);
  float* mcnt = msum + (size_t)NUM_MOTIFS * HDIM;
  float* gcnt = mcnt + NUM_MOTIFS;
  _Float16* gfeat16 = (_Float16*)alloc((size_t)NUM_GRAPHS * HDIM * 2);
  _Float16* hsub16  = (_Float16*)alloc((size_t)(NUM_MOTIFS - 1) * HDIM * 2);
  _Float16* t1g = (_Float16*)alloc((size_t)NUM_GRAPHS * FFN * 2);
  _Float16* t2g = (_Float16*)alloc((size_t)NUM_GRAPHS * FFN * 2);
  _Float16* t1s = (_Float16*)alloc((size_t)(NUM_MOTIFS - 1) * FFN * 2);
  _Float16* t2s = (_Float16*)alloc((size_t)(NUM_MOTIFS - 1) * FFN * 2);

  float* outp = (float*)d_out;
  float* out_gf = outp;                                      // [1500,128]
  float* out_gl = outp + (size_t)NUM_GRAPHS * HDIM;          // [1500,256]
  float* out_sb = out_gl + (size_t)NUM_GRAPHS * (FFN / 2);   // [6000,256]

  const int TPB = 256;
  // ---- edge grouping ----
  launch_zero((float*)ecnt, 130, stream);
  k_hist<<<(E + TPB - 1) / TPB, TPB, 0, stream>>>(ety, ecnt, E);
  k_scan_tiles<<<1, 1, 0, stream>>>(ecnt, eoff, tileR, tileO, ntile);
  k_scatter_edges<<<(E + TPB - 1) / TPB, TPB, 0, stream>>>(src, dst, ety, eoff,
                                                           ecnt + 65, esrc, edst, E);
  // ---- weight / feature conversion ----
  k_cvt<<<((size_t)N * HDIM + TPB - 1) / TPB, TPB, 0, stream>>>(node_feats, fbuf, N * HDIM);
  int wtot = NUM_RELS * HDIM * HDIM;
  k_tcvt<<<(wtot + TPB - 1) / TPB, TPB, 0, stream>>>(W1, Wt1, NUM_RELS, HDIM, HDIM);
  k_tcvt<<<(wtot + TPB - 1) / TPB, TPB, 0, stream>>>(W2, Wt2, NUM_RELS, HDIM, HDIM);
  k_tcvt<<<(HDIM * HDIM + TPB - 1) / TPB, TPB, 0, stream>>>(loopW1, loopT1, 1, HDIM, HDIM);
  k_tcvt<<<(HDIM * HDIM + TPB - 1) / TPB, TPB, 0, stream>>>(loopW2, loopT2, 1, HDIM, HDIM);
  k_cvt<<<(HDIM * HDIM + TPB - 1) / TPB, TPB, 0, stream>>>(resW1, resH1, HDIM * HDIM);
  k_cvt<<<(HDIM * HDIM + TPB - 1) / TPB, TPB, 0, stream>>>(resW2, resH2, HDIM * HDIM);
  k_cvt<<<(FFN * HDIM + TPB - 1) / TPB, TPB, 0, stream>>>(feat_W, featWh, FFN * HDIM);
  k_cvt<<<(FFN * FFN + TPB - 1) / TPB, TPB, 0, stream>>>(mlp_W1, mlpW1h, FFN * FFN);
  k_cvt<<<((FFN / 2) * FFN + TPB - 1) / TPB, TPB, 0, stream>>>(mlp_W2, mlpW2h, (FFN / 2) * FFN);

  // ---- two RGCN layers ----
  const _Float16* Wts[2]   = {Wt1, Wt2};
  const _Float16* loops[2] = {loopT1, loopT2};
  const _Float16* ress[2]  = {resH1, resH2};
  const float* bs[2]  = {b1, b2};
  const float* rbs[2] = {resb1, resb2};
  const float* gs[2]  = {g1, g2};
  const float* bes[2] = {be1, be2};
  for (int l = 0; l < 2; ++l) {
    launch_zero(buf, (size_t)N * HDIM, stream);
    launch_zero(bnsum, 256, stream);
    k_msg<<<MAX_TILES, 32, 0, stream>>>(fbuf, Wts[l], esrc, edst, eoff, tileR,
                                        tileO, ntile, buf);
    k_node<<<N / 16, 32, 0, stream>>>(fbuf, loops[l], ress[l], bs[l], rbs[l],
                                      buf, bnsum);
    k_bnfin<<<1, 128, 0, stream>>>(bnsum, gs[l], bes[l], bnsc, N);
    k_bnapply<<<((size_t)N * HDIM + TPB - 1) / TPB, TPB, 0, stream>>>(buf, bnsc,
                                                                      fbuf, N * HDIM);
  }

  // ---- readouts ----
  launch_zero(out_gf, (size_t)NUM_GRAPHS * HDIM, stream);
  launch_zero(msum, msum_elts, stream);
  k_segscat<<<((size_t)N * 32 + TPB - 1) / TPB, TPB, 0, stream>>>(
      buf, gid, mid, out_gf, gcnt, msum, mcnt, N);
  k_segdiv<<<(NUM_GRAPHS * HDIM + TPB - 1) / TPB, TPB, 0, stream>>>(
      out_gf, gcnt, NUM_GRAPHS, 0, 1, gfeat16);
  k_segdiv<<<((NUM_MOTIFS - 1) * HDIM + TPB - 1) / TPB, TPB, 0, stream>>>(
      msum, mcnt, NUM_MOTIFS - 1, 1, 0, hsub16);

  // ---- heads: 128 -> 512 -> relu 512 -> 256 ----
  dim3 g1a((NUM_GRAPHS + 15) / 16, FFN / 16);
  k_gemm<<<g1a, 32, 0, stream>>>(gfeat16, featWh, feat_b, nullptr, t1g,
                                 NUM_GRAPHS, HDIM, FFN, 0);
  dim3 g1b((NUM_GRAPHS + 15) / 16, FFN / 16);
  k_gemm<<<g1b, 32, 0, stream>>>(t1g, mlpW1h, mlp_b1, nullptr, t2g,
                                 NUM_GRAPHS, FFN, FFN, 1);
  dim3 g1c((NUM_GRAPHS + 15) / 16, (FFN / 2) / 16);
  k_gemm<<<g1c, 32, 0, stream>>>(t2g, mlpW2h, mlp_b2, out_gl, nullptr,
                                 NUM_GRAPHS, FFN, FFN / 2, 0);

  const int MS = NUM_MOTIFS - 1;  // 6000
  dim3 g2a((MS + 15) / 16, FFN / 16);
  k_gemm<<<g2a, 32, 0, stream>>>(hsub16, featWh, feat_b, nullptr, t1s,
                                 MS, HDIM, FFN, 0);
  dim3 g2b((MS + 15) / 16, FFN / 16);
  k_gemm<<<g2b, 32, 0, stream>>>(t1s, mlpW1h, mlp_b1, nullptr, t2s,
                                 MS, FFN, FFN, 1);
  dim3 g2c((MS + 15) / 16, (FFN / 2) / 16);
  k_gemm<<<g2c, 32, 0, stream>>>(t2s, mlpW2h, mlp_b2, out_sb, nullptr,
                                 MS, FFN, FFN / 2, 0);
}
